// HybridGAT_46669114638662
// MI455X (gfx1250) — matrix operand order repaired and checked by reading, not compiled, not run on hardware
//
#include <hip/hip_runtime.h>
#include <hip/hip_bf16.h>
#include <math.h>

#define N_NODES   20000
#define N_EDGES   320000
#define N_GRAPHS  128
#define NODE_IN   128
#define HYB_IN    256
#define HID       128
#define HEADS     4
#define NUM_CLASSES 32
#define E_TOT     (N_EDGES + N_NODES)
#define BN_SCALE  (1.0f / sqrtf(1.0f + 1e-5f))

typedef __attribute__((ext_vector_type(16))) __bf16 v16bf;
typedef __attribute__((ext_vector_type(8)))  __bf16 v8bf;
typedef __attribute__((ext_vector_type(8)))  float  v8f;

// ---------------------------------------------------------------------------
// WMMA GEMM:  C[M,N] = A[M,K] @ W[N,K]^T (+ bias[n]);  A,W in bf16, C f32.
// One wave32 computes a 16 x (NT*16) strip; the A fragment is reused across
// NT v_wmma_f32_16x16x32_bf16. All NT B fragments are loaded (live) before
// the WMMA chain so loads pipeline instead of wait-0 serializing.
// All shapes divide the tile exactly (M=20000=1250*16; N in {512,128,32}).
// ---------------------------------------------------------------------------
template <int NT>
__global__ void gemm_wmma(const __bf16* __restrict__ A, const __bf16* __restrict__ W,
                          const float* __restrict__ bias, float* __restrict__ C,
                          int M, int N, int K) {
    const int lane = threadIdx.x & 31;
    const int wave = (int)(blockIdx.x * (blockDim.x >> 5) + (threadIdx.x >> 5));
    const int groupsN = (N >> 4) / NT;
    const int tM  = (wave / groupsN) << 4;
    const int tN0 = (wave % groupsN) * (NT * 16);
    if (tM >= M) return;

    const int r  = lane & 15;   // A row / B,C column within tile
    const int hh = lane >> 4;   // K-half selector

    const __bf16* __restrict__ arow = A + (size_t)(tM + r) * K;
    const __bf16* wrow[NT];
#pragma unroll
    for (int j = 0; j < NT; ++j)
        wrow[j] = W + (size_t)(tN0 + j * 16 + r) * K;

    v8f acc[NT];
#pragma unroll
    for (int j = 0; j < NT; ++j) acc[j] = (v8f){};

    for (int k0 = 0; k0 < K; k0 += 32) {
        // A 16x32 fragment: per lane K = hh*16 + 0..15 contiguous (2 x b128)
        v16bf a = *(const v16bf*)(arow + k0 + hh * 16);
        // B 32x16 fragments: per lane K = hh*16 + 0..15 contiguous (2 x b128 each)
        v16bf bf[NT];
#pragma unroll
        for (int j = 0; j < NT; ++j)
            bf[j] = *(const v16bf*)(wrow[j] + k0 + hh * 16);
#pragma unroll
        for (int j = 0; j < NT; ++j)
            acc[j] = __builtin_amdgcn_wmma_f32_16x16x32_bf16(
                false, a, false, bf[j], (short)0, acc[j], false, false);
    }

#pragma unroll
    for (int j = 0; j < NT; ++j) {
        const int n = tN0 + j * 16 + r;
        const float bv = bias ? bias[n] : 0.0f;   // one load per column
#pragma unroll
        for (int i = 0; i < 8; ++i)
            C[(size_t)(tM + i + hh * 8) * N + n] = acc[j][i] + bv;
    }
}

// ---------------------------------------------------------------------------
// Elementwise helpers
// ---------------------------------------------------------------------------
__global__ void cvt_bf16(const float* __restrict__ in, __bf16* __restrict__ out, long n) {
    long i = (long)blockIdx.x * blockDim.x + threadIdx.x;
    if (i < n) out[i] = (__bf16)in[i];
}
__global__ void fill_f32(float* p, float v, long n) {
    long i = (long)blockIdx.x * blockDim.x + threadIdx.x;
    if (i < n) p[i] = v;
}
__global__ void fill_u32(unsigned* p, unsigned v, long n) {
    long i = (long)blockIdx.x * blockDim.x + threadIdx.x;
    if (i < n) p[i] = v;
}

// ---------------------------------------------------------------------------
// Attention: per-(node,head) src/dst logits  al = <h[n,h,:], a[h,:]>
// ---------------------------------------------------------------------------
__global__ void attn_logits(const float* __restrict__ H, const float* __restrict__ a_s,
                            const float* __restrict__ a_d, float* __restrict__ als,
                            float* __restrict__ ald, int N, int Hh) {
    int i = blockIdx.x * blockDim.x + threadIdx.x;
    if (i >= N * Hh) return;
    int n = i / Hh, h = i % Hh;
    const float4* row = (const float4*)(H + (size_t)n * Hh * HID + (size_t)h * HID);
    const float4* v1  = (const float4*)(a_s + h * HID);
    const float4* v2  = (const float4*)(a_d + h * HID);
    float s = 0.f, d = 0.f;
#pragma unroll 4
    for (int c = 0; c < HID / 4; ++c) {
        float4 hv = row[c], w1 = v1[c], w2 = v2[c];
        s += hv.x * w1.x + hv.y * w1.y + hv.z * w1.z + hv.w * w1.w;
        d += hv.x * w2.x + hv.y * w2.y + hv.z * w2.z + hv.w * w2.w;
    }
    als[i] = s; ald[i] = d;
}

// monotonic float<->uint encoding for atomicMax on signed floats
__device__ __forceinline__ unsigned enc_f(float f) {
    unsigned u = __float_as_uint(f);
    return (u & 0x80000000u) ? ~u : (u | 0x80000000u);
}
__device__ __forceinline__ float dec_f(unsigned u) {
    return __uint_as_float((u & 0x80000000u) ? (u & 0x7FFFFFFFu) : ~u);
}

// leaky-relu logits + segment max over dst (self-loops appended implicitly)
__global__ void edge_logit_max(const int* __restrict__ ei, const float* __restrict__ als,
                               const float* __restrict__ ald, float* __restrict__ elog,
                               unsigned* __restrict__ mx, int Hh) {
    long i = (long)blockIdx.x * blockDim.x + threadIdx.x;
    if (i >= (long)E_TOT * Hh) return;
    int e = (int)(i / Hh), h = (int)(i % Hh);
    int s = (e < N_EDGES) ? ei[e]           : (e - N_EDGES);
    int d = (e < N_EDGES) ? ei[N_EDGES + e] : (e - N_EDGES);
    float l = als[(size_t)s * Hh + h] + ald[(size_t)d * Hh + h];
    l = (l >= 0.f) ? l : 0.2f * l;
    elog[i] = l;
    atomicMax(&mx[(size_t)d * Hh + h], enc_f(l));
}

// exp(logit - max[dst]) + segment sum of denominators
__global__ void edge_exp_sum(const int* __restrict__ ei, float* __restrict__ elog,
                             const unsigned* __restrict__ mx, float* __restrict__ denom,
                             int Hh) {
    long i = (long)blockIdx.x * blockDim.x + threadIdx.x;
    if (i >= (long)E_TOT * Hh) return;
    int e = (int)(i / Hh), h = (int)(i % Hh);
    int d = (e < N_EDGES) ? ei[N_EDGES + e] : (e - N_EDGES);
    float m  = dec_f(mx[(size_t)d * Hh + h]);
    float ex = __expf(elog[i] - m);
    elog[i] = ex;
    atomicAdd(&denom[(size_t)d * Hh + h], ex);
}

// out[dst] += h[src] * alpha   (one block per edge, lanes over channels)
__global__ void scatter_msg(const float* __restrict__ H, const float* __restrict__ elog,
                            const float* __restrict__ denom, const int* __restrict__ ei,
                            float* __restrict__ out, int Hh, int C) {
    int e = blockIdx.x;
    int s = (e < N_EDGES) ? ei[e]           : (e - N_EDGES);
    int d = (e < N_EDGES) ? ei[N_EDGES + e] : (e - N_EDGES);
    int F = Hh * C;
    for (int ch = threadIdx.x; ch < F; ch += blockDim.x) {
        int h = ch / C;
        float alpha = elog[(size_t)e * Hh + h] / (denom[(size_t)d * Hh + h] + 1e-16f);
        atomicAdd(&out[(size_t)d * F + ch], H[(size_t)s * F + ch] * alpha);
    }
}

// (+bias) -> eval-BN -> activation; OutT = float or __bf16 (feeds next GEMM)
// act: 0=none, 1=relu, 2=elu
template <typename OutT>
__global__ void bn_act(const float* __restrict__ in, OutT* __restrict__ out,
                       const float* __restrict__ bias, const float* __restrict__ g,
                       const float* __restrict__ be, int N, int F,
                       int outStride, int outColOff, int act) {
    long i = (long)blockIdx.x * blockDim.x + threadIdx.x;
    if (i >= (long)N * F) return;
    int n = (int)(i / F), c = (int)(i % F);
    float v = in[i];
    if (bias) v += bias[c];
    if (g)    v = v * (g[c] * BN_SCALE) + be[c];
    if (act == 1)      v = fmaxf(v, 0.f);
    else if (act == 2) v = (v > 0.f) ? v : expm1f(v);
    out[(size_t)n * outStride + outColOff + c] = (OutT)v;
}

// global mean pool: per-graph sums + counts
__global__ void pool_sum(const float* __restrict__ H, const int* __restrict__ batch,
                         float* __restrict__ sums, float* __restrict__ cnts) {
    int n = blockIdx.x;
    int b = batch[n];
    for (int c = threadIdx.x; c < HID; c += blockDim.x)
        atomicAdd(&sums[(size_t)b * HID + c], H[(size_t)n * HID + c]);
    if (threadIdx.x == 0) atomicAdd(&cnts[b], 1.0f);
}
__global__ void pool_div(const float* __restrict__ sums, const float* __restrict__ cnts,
                         __bf16* __restrict__ comb) {
    int i = blockIdx.x * blockDim.x + threadIdx.x;
    if (i >= N_GRAPHS * HID) return;
    int g = i / HID, c = i % HID;
    comb[(size_t)g * (2 * HID) + c] = (__bf16)(sums[i] / fmaxf(cnts[g], 1.0f));
}

// ---------------------------------------------------------------------------
static inline void launch_gemm(const __bf16* A, const __bf16* W, const float* bias,
                               float* C, int M, int N, int K, hipStream_t s) {
    int tilesN = N / 16;
    if ((tilesN & 3) == 0) {
        int waves = (M / 16) * (tilesN / 4);
        gemm_wmma<4><<<(waves + 3) / 4, 128, 0, s>>>(A, W, bias, C, M, N, K);
    } else {
        int waves = (M / 16) * (tilesN / 2);
        gemm_wmma<2><<<(waves + 3) / 4, 128, 0, s>>>(A, W, bias, C, M, N, K);
    }
}
static inline int cdiv(long n, int b) { return (int)((n + b - 1) / b); }

extern "C" void kernel_launch(void* const* d_in, const int* in_sizes, int n_in,
                              void* d_out, int out_size, void* d_ws, size_t ws_size,
                              hipStream_t stream) {
    const float* x    = (const float*)d_in[0];
    const int*   ei   = (const int*)  d_in[1];
    const int*   bat  = (const int*)  d_in[2];
    const float* hyb  = (const float*)d_in[3];
    const float* W1   = (const float*)d_in[4];
    const float* as1  = (const float*)d_in[5];
    const float* ad1  = (const float*)d_in[6];
    const float* b1   = (const float*)d_in[7];
    const float* W2   = (const float*)d_in[8];
    const float* as2  = (const float*)d_in[9];
    const float* ad2  = (const float*)d_in[10];
    const float* b2   = (const float*)d_in[11];
    const float* g_bn1  = (const float*)d_in[12];
    const float* be_bn1 = (const float*)d_in[13];
    const float* g_bn2  = (const float*)d_in[14];
    const float* be_bn2 = (const float*)d_in[15];
    const float* g_hbn1 = (const float*)d_in[16];
    const float* be_hbn1= (const float*)d_in[17];
    const float* g_hbn2 = (const float*)d_in[18];
    const float* be_hbn2= (const float*)d_in[19];
    const float* g_fbn  = (const float*)d_in[20];
    const float* be_fbn = (const float*)d_in[21];
    const float* hW1 = (const float*)d_in[22];
    const float* hb1 = (const float*)d_in[23];
    const float* hW2 = (const float*)d_in[24];
    const float* hb2 = (const float*)d_in[25];
    const float* fW  = (const float*)d_in[26];
    const float* fb  = (const float*)d_in[27];
    const float* cW  = (const float*)d_in[28];
    const float* cb  = (const float*)d_in[29];
    float* out = (float*)d_out;

    // ---- workspace bump allocator (256B aligned) ----
    char* ws = (char*)d_ws;
    size_t off = 0;
    auto alloc = [&](size_t bytes) -> char* {
        char* p = ws + off;
        off += (bytes + 255) & ~(size_t)255;
        return p;
    };
    float*    bufA  = (float*)   alloc((size_t)N_NODES * HEADS * HID * 4); // 41 MB
    float*    bufB  = (float*)   alloc((size_t)N_NODES * HEADS * HID * 4); // 41 MB
    __bf16*   xb    = (__bf16*)  alloc((size_t)N_NODES * NODE_IN * 2);     // 5.1 MB
    __bf16*   hybb  = (__bf16*)  alloc((size_t)N_GRAPHS * HYB_IN * 2);
    __bf16*   W1b   = (__bf16*)  alloc((size_t)HEADS * HID * NODE_IN * 2);
    __bf16*   W2b   = (__bf16*)  alloc((size_t)HID * HEADS * HID * 2);
    __bf16*   hW1b  = (__bf16*)  alloc((size_t)HID * HYB_IN * 2);
    __bf16*   hW2b  = (__bf16*)  alloc((size_t)HID * HID * 2);
    __bf16*   fWb   = (__bf16*)  alloc((size_t)HID * 2 * HID * 2);
    __bf16*   cWb   = (__bf16*)  alloc((size_t)NUM_CLASSES * HID * 2);
    float*    als1  = (float*)   alloc((size_t)N_NODES * HEADS * 4);
    float*    ald1  = (float*)   alloc((size_t)N_NODES * HEADS * 4);
    float*    elog1 = (float*)   alloc((size_t)E_TOT   * HEADS * 4);
    unsigned* mx1   = (unsigned*)alloc((size_t)N_NODES * HEADS * 4);
    float*    den1  = (float*)   alloc((size_t)N_NODES * HEADS * 4);
    float*    als2  = (float*)   alloc((size_t)N_NODES * 4);
    float*    ald2  = (float*)   alloc((size_t)N_NODES * 4);
    float*    elog2 = (float*)   alloc((size_t)E_TOT   * 4);
    unsigned* mx2   = (unsigned*)alloc((size_t)N_NODES * 4);
    float*    den2  = (float*)   alloc((size_t)N_NODES * 4);
    float*    sums  = (float*)   alloc((size_t)N_GRAPHS * HID * 4);
    float*    cnts  = (float*)   alloc((size_t)N_GRAPHS * 4);
    float*    t1    = (float*)   alloc((size_t)N_GRAPHS * HID * 4);
    __bf16*   t1b   = (__bf16*)  alloc((size_t)N_GRAPHS * HID * 2);
    float*    t2    = (float*)   alloc((size_t)N_GRAPHS * HID * 4);
    __bf16*   comb  = (__bf16*)  alloc((size_t)N_GRAPHS * 2 * HID * 2);
    float*    fus   = (float*)   alloc((size_t)N_GRAPHS * HID * 4);
    __bf16*   fusb  = (__bf16*)  alloc((size_t)N_GRAPHS * HID * 2);
    // bf16 activations after GAT1 epilogue alias dead halves of bufA
    __bf16*   actAb = (__bf16*)bufA;                                // 20.5 MB alias
    float*    agg2  = bufA + (size_t)N_NODES * HEADS * HID / 2;    // upper half alias
    (void)ws_size;

    const int TB = 256;

    // ---- one-time (per call) bf16 conversions of GEMM operands ----
    cvt_bf16<<<cdiv((long)N_NODES * NODE_IN, TB), TB, 0, stream>>>(x,   xb,   (long)N_NODES * NODE_IN);
    cvt_bf16<<<cdiv((long)N_GRAPHS * HYB_IN, TB), TB, 0, stream>>>(hyb, hybb, (long)N_GRAPHS * HYB_IN);
    cvt_bf16<<<cdiv((long)HEADS * HID * NODE_IN, TB), TB, 0, stream>>>(W1,  W1b,  (long)HEADS * HID * NODE_IN);
    cvt_bf16<<<cdiv((long)HID * HEADS * HID, TB), TB, 0, stream>>>(W2,  W2b,  (long)HID * HEADS * HID);
    cvt_bf16<<<cdiv((long)HID * HYB_IN, TB), TB, 0, stream>>>(hW1, hW1b, (long)HID * HYB_IN);
    cvt_bf16<<<cdiv((long)HID * HID, TB), TB, 0, stream>>>(hW2, hW2b, (long)HID * HID);
    cvt_bf16<<<cdiv((long)HID * 2 * HID, TB), TB, 0, stream>>>(fW,  fWb,  (long)HID * 2 * HID);
    cvt_bf16<<<cdiv((long)NUM_CLASSES * HID, TB), TB, 0, stream>>>(cW,  cWb,  (long)NUM_CLASSES * HID);

    // ================= GAT layer 1 (heads=4, F=512) =================
    launch_gemm(xb, W1b, nullptr, bufA, N_NODES, HEADS * HID, NODE_IN, stream);
    attn_logits<<<cdiv((long)N_NODES * HEADS, TB), TB, 0, stream>>>(bufA, as1, ad1, als1, ald1, N_NODES, HEADS);
    fill_u32<<<cdiv((long)N_NODES * HEADS, TB), TB, 0, stream>>>(mx1, 0u, (long)N_NODES * HEADS);
    fill_f32<<<cdiv((long)N_NODES * HEADS, TB), TB, 0, stream>>>(den1, 0.f, (long)N_NODES * HEADS);
    fill_f32<<<cdiv((long)N_NODES * HEADS * HID, TB), TB, 0, stream>>>(bufB, 0.f, (long)N_NODES * HEADS * HID);
    edge_logit_max<<<cdiv((long)E_TOT * HEADS, TB), TB, 0, stream>>>(ei, als1, ald1, elog1, mx1, HEADS);
    edge_exp_sum <<<cdiv((long)E_TOT * HEADS, TB), TB, 0, stream>>>(ei, elog1, mx1, den1, HEADS);
    scatter_msg<<<E_TOT, 128, 0, stream>>>(bufA, elog1, den1, ei, bufB, HEADS, HID);
    // +b1, BN1, ELU -> bf16 activations (alias over dead bufA)
    bn_act<__bf16><<<cdiv((long)N_NODES * HEADS * HID, TB), TB, 0, stream>>>(
        bufB, actAb, b1, g_bn1, be_bn1, N_NODES, HEADS * HID, HEADS * HID, 0, 2);

    // ================= GAT layer 2 (heads=1, F=128) =================
    launch_gemm(actAb, W2b, nullptr, bufB, N_NODES, HID, HEADS * HID, stream);
    attn_logits<<<cdiv((long)N_NODES, TB), TB, 0, stream>>>(bufB, as2, ad2, als2, ald2, N_NODES, 1);
    fill_u32<<<cdiv((long)N_NODES, TB), TB, 0, stream>>>(mx2, 0u, (long)N_NODES);
    fill_f32<<<cdiv((long)N_NODES, TB), TB, 0, stream>>>(den2, 0.f, (long)N_NODES);
    fill_f32<<<cdiv((long)N_NODES * HID, TB), TB, 0, stream>>>(agg2, 0.f, (long)N_NODES * HID);
    edge_logit_max<<<cdiv((long)E_TOT, TB), TB, 0, stream>>>(ei, als2, ald2, elog2, mx2, 1);
    edge_exp_sum <<<cdiv((long)E_TOT, TB), TB, 0, stream>>>(ei, elog2, mx2, den2, 1);
    scatter_msg<<<E_TOT, 128, 0, stream>>>(bufB, elog2, den2, ei, agg2, 1, HID);
    // +b2, BN2, ELU (in place, f32 for pooling)
    bn_act<float><<<cdiv((long)N_NODES * HID, TB), TB, 0, stream>>>(
        agg2, agg2, b2, g_bn2, be_bn2, N_NODES, HID, HID, 0, 2);

    // ================= global mean pool -> comb[:, 0:128] (bf16) ============
    fill_f32<<<cdiv((long)N_GRAPHS * HID, TB), TB, 0, stream>>>(sums, 0.f, (long)N_GRAPHS * HID);
    fill_f32<<<1, N_GRAPHS, 0, stream>>>(cnts, 0.f, (long)N_GRAPHS);
    pool_sum<<<N_NODES, 128, 0, stream>>>(agg2, bat, sums, cnts);
    pool_div<<<cdiv((long)N_GRAPHS * HID, TB), TB, 0, stream>>>(sums, cnts, comb);

    // ================= hybrid MLP -> comb[:, 128:256] (bf16) ================
    launch_gemm(hybb, hW1b, hb1, t1, N_GRAPHS, HID, HYB_IN, stream);
    bn_act<__bf16><<<cdiv((long)N_GRAPHS * HID, TB), TB, 0, stream>>>(
        t1, t1b, nullptr, g_hbn1, be_hbn1, N_GRAPHS, HID, HID, 0, 1);
    launch_gemm(t1b, hW2b, hb2, t2, N_GRAPHS, HID, HID, stream);
    bn_act<__bf16><<<cdiv((long)N_GRAPHS * HID, TB), TB, 0, stream>>>(
        t2, comb, nullptr, g_hbn2, be_hbn2, N_GRAPHS, HID, 2 * HID, HID, 1);

    // ================= fusion + classifier =================
    launch_gemm(comb, fWb, fb, fus, N_GRAPHS, HID, 2 * HID, stream);
    bn_act<__bf16><<<cdiv((long)N_GRAPHS * HID, TB), TB, 0, stream>>>(
        fus, fusb, nullptr, g_fbn, be_fbn, N_GRAPHS, HID, HID, 0, 1);
    launch_gemm(fusb, cWb, cb, out, N_GRAPHS, NUM_CLASSES, HID, stream);
    (void)out_size; (void)n_in; (void)in_sizes;
}
